// Qwen3VLVisionAttention_5463198400898
// MI455X (gfx1250) — matrix operand order repaired and checked by reading, not compiled, not run on hardware
//
#include <hip/hip_runtime.h>
#include <hip/hip_bf16.h>

// ---------------- problem constants ----------------
#define TT   4096
#define DD   1280
#define HH   16
#define DH   80
#define DP   96          // Dh padded to 3 K-steps of 32 for WMMA

typedef __attribute__((ext_vector_type(16))) __bf16 v16bf;
typedef __attribute__((ext_vector_type(8)))  __bf16 v8bf;
typedef __attribute__((ext_vector_type(8)))  float  v8f;

#define DEV static __device__ __forceinline__

DEV __bf16 f2bf(float f) {
  unsigned u = __builtin_bit_cast(unsigned, f);
  u += 0x7FFFu + ((u >> 16) & 1u);            // round-to-nearest-even
  unsigned short h = (unsigned short)(u >> 16);
  return __builtin_bit_cast(__bf16, h);
}
DEV float bf2f(__bf16 b) {
  unsigned short h = __builtin_bit_cast(unsigned short, b);
  unsigned u = ((unsigned)h) << 16;
  return __builtin_bit_cast(float, u);
}
DEV v8f v8f_zero() {
  v8f z;
#pragma unroll
  for (int i = 0; i < 8; ++i) z[i] = 0.0f;
  return z;
}

union Frag16 { v16bf v; v8bf h[2]; };

// ---- CDNA5 async global->LDS copy (ASYNCcnt path, CDNA5 ISA) ----
DEV void async_copy_b128(unsigned lds_off, const void* g) {
  unsigned long long ga = (unsigned long long)(uintptr_t)g;
  asm volatile("global_load_async_to_lds_b128 %0, %1, off"
               :: "v"(lds_off), "v"(ga) : "memory");
}
DEV void wait_async_le6() {
#if __has_builtin(__builtin_amdgcn_s_wait_asynccnt)
  __builtin_amdgcn_s_wait_asynccnt(6);
#else
  asm volatile("s_wait_asynccnt 6" ::: "memory");
#endif
}
DEV void wait_async_le0() {
#if __has_builtin(__builtin_amdgcn_s_wait_asynccnt)
  __builtin_amdgcn_s_wait_asynccnt(0);
#else
  asm volatile("s_wait_asynccnt 0" ::: "memory");
#endif
}

// ---------------- conversion kernels ----------------
__global__ void cvt_copy_f32_bf16(const float* __restrict__ in,
                                  __bf16* __restrict__ out, int n) {
  int i = blockIdx.x * blockDim.x + threadIdx.x;
  if (i < n) out[i] = f2bf(in[i]);
}

// in[rows][cols] (f32) -> out[cols][rows] (bf16)
__global__ void cvt_transpose_f32_bf16(const float* __restrict__ in,
                                       __bf16* __restrict__ out,
                                       int rows, int cols) {
  int i = blockIdx.x * blockDim.x + threadIdx.x;
  if (i >= rows * cols) return;
  int r = i / cols, c = i % cols;
  out[(size_t)c * rows + r] = f2bf(in[(size_t)r * cols + c]);
}

// ---------------- WMMA GEMM: C[M,N] = A[M,K] * Bt[N,K]^T + bias ----------------
// block: 256 threads = 8 waves; block tile 128x128; wave tile 64x32.
// A/W are L2-resident (192MB L2), so direct b128 fragment streaming is at the
// L2 roofline; the compiler pipelines loads against WMMAs (seen in round-1 asm).
template <bool OUT_BF16>
__global__ __launch_bounds__(256)
void gemm_bf16_wmma(const __bf16* __restrict__ A, const __bf16* __restrict__ Bt,
                    const float* __restrict__ bias, void* __restrict__ outp,
                    int M, int N, int K) {
  const int l  = threadIdx.x & 31;
  const int w  = threadIdx.x >> 5;
  const int lr = l & 15;              // row/col selector within fragment
  const int lg = l >> 4;              // half-wave group
  const int wm = blockIdx.y * 128 + (w >> 2) * 64;
  const int wn = blockIdx.x * 128 + (w & 3) * 32;

  v8f acc[4][2];
#pragma unroll
  for (int f = 0; f < 4; ++f)
#pragma unroll
    for (int g = 0; g < 2; ++g) acc[f][g] = v8f_zero();

  for (int k0 = 0; k0 < K; k0 += 32) {
    Frag16 a[4], b[2];
#pragma unroll
    for (int f = 0; f < 4; ++f) {
      const __bf16* p = A + (size_t)(wm + f * 16 + lr) * K + k0 + lg * 8;
      a[f].h[0] = *(const v8bf*)(p);
      a[f].h[1] = *(const v8bf*)(p + 16);
      __builtin_prefetch(p + 32, 0, 1);       // next K tile -> global_prefetch
    }
#pragma unroll
    for (int g = 0; g < 2; ++g) {
      const __bf16* p = Bt + (size_t)(wn + g * 16 + lr) * K + k0 + lg * 8;
      b[g].h[0] = *(const v8bf*)(p);
      b[g].h[1] = *(const v8bf*)(p + 16);
      __builtin_prefetch(p + 32, 0, 1);
    }
#pragma unroll
    for (int f = 0; f < 4; ++f)
#pragma unroll
      for (int g = 0; g < 2; ++g)
        acc[f][g] = __builtin_amdgcn_wmma_f32_16x16x32_bf16(
            false, a[f].v, false, b[g].v, (short)0, acc[f][g], false, false);
  }

#pragma unroll
  for (int g = 0; g < 2; ++g) {
    const int col = wn + g * 16 + lr;
    const float bv = bias[col];
#pragma unroll
    for (int f = 0; f < 4; ++f) {
#pragma unroll
      for (int j = 0; j < 8; ++j) {
        const int row = wm + f * 16 + lg * 8 + j;
        float v = acc[f][g][j] + bv;
        if constexpr (OUT_BF16)
          ((__bf16*)outp)[(size_t)row * N + col] = f2bf(v);
        else
          ((float*)outp)[(size_t)row * N + col] = v;
      }
    }
  }
}

// ---------------- RoPE + repack ----------------
// qkv: [T][3D] bf16  ->  Q,K: [H][T][DP] bf16 (zero-padded), Vt: [H][DH][T] bf16
__global__ void rope_pack(const __bf16* __restrict__ qkv,
                          const float* __restrict__ rope,   // [T][DH/2]
                          __bf16* __restrict__ Qo, __bf16* __restrict__ Ko,
                          __bf16* __restrict__ Vt) {
  int i = blockIdx.x * blockDim.x + threadIdx.x;
  if (i >= TT * HH * DP) return;
  int d = i % DP;
  int h = (i / DP) % HH;
  int t = i / (DP * HH);
  size_t qki = (size_t)h * TT * DP + (size_t)t * DP + d;
  if (d >= DH) { Qo[qki] = f2bf(0.0f); Ko[qki] = f2bf(0.0f); return; }
  size_t base = (size_t)t * (3 * DD) + h * DH + d;
  float qv = bf2f(qkv[base]);
  float kv = bf2f(qkv[base + DD]);
  float vv = bf2f(qkv[base + 2 * DD]);
  int dr = (d < DH / 2) ? d : d - DH / 2;
  float ang = rope[(size_t)t * (DH / 2) + dr];
  float c = __cosf(ang), s = __sinf(ang);
  float qr, kr;
  if (d < DH / 2) { qr = -bf2f(qkv[base + DH / 2]); kr = -bf2f(qkv[base + DD + DH / 2]); }
  else            { qr =  bf2f(qkv[base - DH / 2]); kr =  bf2f(qkv[base + DD - DH / 2]); }
  Qo[qki] = f2bf(qv * c + qr * s);
  Ko[qki] = f2bf(kv * c + kr * s);
  Vt[(size_t)h * DH * TT + (size_t)d * TT + t] = f2bf(vv);
}

// ---------------- flash attention with WMMA + async-LDS K/V staging ----------------
// grid: (T/128, H); 256 threads = 8 waves; each wave owns 16 q rows.
// K/V tiles (64 keys) are staged once per block into LDS with
// global_load_async_to_lds_b128 (double buffered), shared by all 8 waves
// (8x reduction in L2 traffic vs per-wave streaming).
__global__ __launch_bounds__(256)
void attn_wmma(const __bf16* __restrict__ Q,   // [H][T][DP]
               const __bf16* __restrict__ Kb,  // [H][T][DP]
               const __bf16* __restrict__ Vt,  // [H][DH][T]
               const int* __restrict__ cu, int ncu,
               __bf16* __restrict__ Out, float sm_scale) {
  __shared__ __bf16 KtS[2][64][DP];            // 2 x 12288 B
  __shared__ __bf16 VtS[2][DH][64];            // 2 x 10240 B
  __shared__ __bf16 pbuf[8][16][64];           // per-wave P tile (C->A relayout)

  const int tid = threadIdx.x;
  const int l  = tid & 31;
  const int w  = tid >> 5;
  const int lr = l & 15;
  const int lg = l >> 4;
  const int qb = blockIdx.x;
  const int h  = blockIdx.y;
  const int q0 = qb * 128 + w * 16;

  const __bf16* Qh = Q  + (size_t)h * TT * DP;
  const __bf16* Kh = Kb + (size_t)h * TT * DP;
  const __bf16* Vh = Vt + (size_t)h * DH * TT;

  unsigned ktbase[2], vtbase[2];
#pragma unroll
  for (int b = 0; b < 2; ++b) {
    ktbase[b] = (unsigned)(uintptr_t)&KtS[b][0][0];   // low 32 bits = LDS offset
    vtbase[b] = (unsigned)(uintptr_t)&VtS[b][0][0];
  }

  // issue one 64-key tile: K 768 chunks (contiguous rows), V 640 chunks
  // (80 strided rows of 128B). V padded to 768 via wraparound so every thread
  // issues exactly 6 async instructions -> uniform s_wait_asynccnt immediate.
  auto issue_tile = [&](int buf, int kb2) {
    const __bf16* gk = Kh + (size_t)kb2 * DP;
    unsigned kt = ktbase[buf];
#pragma unroll
    for (int c = 0; c < 3; ++c) {
      int chunk = tid + c * 256;                       // 0..767
      async_copy_b128(kt + (unsigned)chunk * 16, gk + chunk * 8);
    }
    unsigned vtb = vtbase[buf];
#pragma unroll
    for (int c = 0; c < 3; ++c) {
      int chunk = tid + c * 256;
      if (chunk >= 640) chunk -= 640;                  // duplicate, same data
      int row = chunk >> 3, cc = chunk & 7;
      async_copy_b128(vtb + (unsigned)(row * 64 + cc * 8) * 2,
                      Vh + (size_t)row * TT + kb2 + cc * 8);
    }
  };

  int cuv[8];
#pragma unroll
  for (int i = 0; i < 8; ++i) cuv[i] = (i < ncu) ? cu[i] : 0x7fffffff;

  int segq[8];
#pragma unroll
  for (int j = 0; j < 8; ++j) {
    int pos = q0 + lg * 8 + j, s = 0;
#pragma unroll
    for (int i = 0; i < 8; ++i) s += (pos >= cuv[i]) ? 1 : 0;
    segq[j] = s;
  }

  // Q fragments for the 3 dh-chunks are invariant over the key loop
  Frag16 qa[3];
#pragma unroll
  for (int c = 0; c < 3; ++c) {
    const __bf16* p = Qh + (size_t)(q0 + lr) * DP + c * 32 + lg * 8;
    qa[c].h[0] = *(const v8bf*)(p);
    qa[c].h[1] = *(const v8bf*)(p + 16);
  }

  v8f o[5];
#pragma unroll
  for (int f = 0; f < 5; ++f) o[f] = v8f_zero();
  float mrow[8], lrow[8];
#pragma unroll
  for (int j = 0; j < 8; ++j) { mrow[j] = -__builtin_inff(); lrow[j] = 0.0f; }

  issue_tile(0, 0);                                    // prologue: tile 0 in flight

  const int NKB = TT / 64;
  for (int kbi = 0; kbi < NKB; ++kbi) {
    const int kb  = kbi * 64;
    const int cur = kbi & 1;

    __syncthreads();              // all waves done reading buffer cur^1 (iter kbi-1)
    if (kbi + 1 < NKB) {
      issue_tile(cur ^ 1, kb + 64);
      wait_async_le6();           // 6 newer in flight => tile `cur` complete (in-order)
    } else {
      wait_async_le0();
    }
    __syncthreads();              // every wave's tile `cur` resident in LDS

    // ---- S = Q * K^T for 16 q rows x 64 keys (K from LDS) ----
    v8f s[4];
#pragma unroll
    for (int f = 0; f < 4; ++f) s[f] = v8f_zero();
#pragma unroll
    for (int c = 0; c < 3; ++c) {
      Frag16 b[4];
#pragma unroll
      for (int f = 0; f < 4; ++f) {
        const __bf16* p = &KtS[cur][f * 16 + lr][c * 32 + lg * 8];
        b[f].h[0] = *(const v8bf*)(p);
        b[f].h[1] = *(const v8bf*)(p + 16);
      }
#pragma unroll
      for (int f = 0; f < 4; ++f)
        s[f] = __builtin_amdgcn_wmma_f32_16x16x32_bf16(
            false, qa[c].v, false, b[f].v, (short)0, s[f], false, false);
    }

    int segk[4];
#pragma unroll
    for (int f = 0; f < 4; ++f) {
      int pos = kb + f * 16 + lr, sg = 0;
#pragma unroll
      for (int i = 0; i < 8; ++i) sg += (pos >= cuv[i]) ? 1 : 0;
      segk[f] = sg;
    }

    // ---- online softmax (row = lg*8+j, spread over 16 lanes of half-wave) ----
#pragma unroll
    for (int j = 0; j < 8; ++j) {
      float vm[4];
      float mloc = -1.0e9f;
#pragma unroll
      for (int f = 0; f < 4; ++f) {
        float v = (segq[j] == segk[f]) ? s[f][j] * sm_scale : -1.0e9f;
        vm[f] = v;
        mloc = fmaxf(mloc, v);
      }
#pragma unroll
      for (int off = 1; off < 16; off <<= 1)
        mloc = fmaxf(mloc, __shfl_xor(mloc, off, 32));
      float mn = fmaxf(mrow[j], mloc);
      float al = __expf(mrow[j] - mn);
      float rs = 0.0f;
#pragma unroll
      for (int f = 0; f < 4; ++f) {
        float pv = (segq[j] == segk[f]) ? __expf(vm[f] - mn) : 0.0f;
        rs += pv;
        pbuf[w][lg * 8 + j][f * 16 + lr] = f2bf(pv);
      }
#pragma unroll
      for (int off = 1; off < 16; off <<= 1)
        rs += __shfl_xor(rs, off, 32);
      lrow[j] = lrow[j] * al + rs;
      mrow[j] = mn;
#pragma unroll
      for (int f = 0; f < 5; ++f) o[f][j] *= al;
    }

    // ---- O += P * V  (P relayout via LDS; V tile from LDS) ----
#pragma unroll
    for (int c = 0; c < 2; ++c) {
      Frag16 pa;
      const __bf16* pp = &pbuf[w][lr][c * 32 + lg * 8];
      pa.h[0] = *(const v8bf*)(pp);
      pa.h[1] = *(const v8bf*)(pp + 16);
#pragma unroll
      for (int f = 0; f < 5; ++f) {
        Frag16 vb;
        const __bf16* vp = &VtS[cur][f * 16 + lr][c * 32 + lg * 8];
        vb.h[0] = *(const v8bf*)(vp);
        vb.h[1] = *(const v8bf*)(vp + 16);
        o[f] = __builtin_amdgcn_wmma_f32_16x16x32_bf16(
            false, pa.v, false, vb.v, (short)0, o[f], false, false);
      }
    }
  }

  // ---- normalize and write [T][H*DH] bf16 ----
#pragma unroll
  for (int j = 0; j < 8; ++j) {
    float inv = 1.0f / lrow[j];
    int row = q0 + lg * 8 + j;
#pragma unroll
    for (int f = 0; f < 5; ++f)
      Out[(size_t)row * DD + h * DH + f * 16 + lr] = f2bf(o[f][j] * inv);
  }
}

// ---------------- launcher ----------------
extern "C" void kernel_launch(void* const* d_in, const int* in_sizes, int n_in,
                              void* d_out, int out_size, void* d_ws, size_t ws_size,
                              hipStream_t stream) {
  const float* x     = (const float*)d_in[0];
  const float* rope  = (const float*)d_in[1];
  const int*   cu    = (const int*)d_in[2];
  const float* Wqkv  = (const float*)d_in[3];
  const float* bqkv  = (const float*)d_in[4];
  const float* Wproj = (const float*)d_in[5];
  const float* bproj = (const float*)d_in[6];
  const int ncu = in_sizes[2];

  char* ws = (char*)d_ws;
  size_t off = 0;
  auto carve = [&](size_t bytes) -> void* {
    void* p = ws + off;
    off += (bytes + 255) & ~(size_t)255;
    return p;
  };
  __bf16* xb     = (__bf16*)carve((size_t)TT * DD * 2);
  __bf16* WqkvT  = (__bf16*)carve((size_t)3 * DD * DD * 2);
  __bf16* WprojT = (__bf16*)carve((size_t)DD * DD * 2);
  __bf16* qkvb   = (__bf16*)carve((size_t)TT * 3 * DD * 2);
  __bf16* Qp     = (__bf16*)carve((size_t)HH * TT * DP * 2);
  __bf16* Kp     = (__bf16*)carve((size_t)HH * TT * DP * 2);
  __bf16* Vt     = (__bf16*)carve((size_t)HH * DH * TT * 2);
  __bf16* AOut   = (__bf16*)carve((size_t)TT * DD * 2);
  (void)ws_size; (void)n_in; (void)out_size;

  // 1) convert inputs to bf16 (weights transposed to Bt[N][K])
  {
    int n = TT * DD;
    cvt_copy_f32_bf16<<<(n + 255) / 256, 256, 0, stream>>>(x, xb, n);
  }
  {
    int n = DD * 3 * DD;
    cvt_transpose_f32_bf16<<<(n + 255) / 256, 256, 0, stream>>>(Wqkv, WqkvT, DD, 3 * DD);
  }
  {
    int n = DD * DD;
    cvt_transpose_f32_bf16<<<(n + 255) / 256, 256, 0, stream>>>(Wproj, WprojT, DD, DD);
  }

  // 2) QKV GEMM (bf16 WMMA, f32 accumulate, bias, bf16 out)
  gemm_bf16_wmma<true><<<dim3(3 * DD / 128, TT / 128), 256, 0, stream>>>(
      xb, WqkvT, bqkv, (void*)qkvb, TT, 3 * DD, DD);

  // 3) RoPE + repack into attention layouts
  {
    int n = TT * HH * DP;
    rope_pack<<<(n + 255) / 256, 256, 0, stream>>>(qkvb, rope, Qp, Kp, Vt);
  }

  // 4) segment-masked flash attention (async-LDS staged K/V)
  {
    float sm_scale = 1.0f / sqrtf((float)DH);
    attn_wmma<<<dim3(TT / 128, HH), 256, 0, stream>>>(Qp, Kp, Vt, cu, ncu, AOut, sm_scale);
  }

  // 5) output projection (f32 out to d_out)
  gemm_bf16_wmma<false><<<dim3(DD / 128, TT / 128), 256, 0, stream>>>(
      AOut, WprojT, bproj, d_out, TT, DD, DD);
}
